// GlueStick_3917010174425
// MI455X (gfx1250) — matrix operand order repaired and checked by reading, not compiled, not run on hardware
//
#include <hip/hip_runtime.h>

typedef __bf16 bf16;
typedef __attribute__((ext_vector_type(16))) __bf16 v16bf;
typedef __attribute__((ext_vector_type(8)))  __bf16 v8bf;
typedef __attribute__((ext_vector_type(2)))  __bf16 v2bf;
typedef __attribute__((ext_vector_type(8)))  float  v8f;

__device__ __forceinline__ v8f wmma_bf16(v16bf a, v16bf b, v8f c) {
  // D = A(16x32 bf16) * B(32x16 bf16) + C(16x16 f32)
  return __builtin_amdgcn_wmma_f32_16x16x32_bf16(false, a, false, b, (short)0, c,
                                                 false, false);
}

__device__ __forceinline__ v8bf ld8(const bf16* p) { return *(const v8bf*)p; }

__device__ __forceinline__ v16bf cat8(v8bf lo, v8bf hi) {
  return __builtin_shufflevector(lo, hi, 0, 1, 2, 3, 4, 5, 6, 7,
                                 8, 9, 10, 11, 12, 13, 14, 15);
}

// ---------------------------------------------------------------------------
// Generic fused GEMM: out = EPI( W[M,K] @ X[K,Nc] + bias )
// XMODE 0: X0 plain, row stride ldx0
// XMODE 1: K-concat of X0 (first Dd rows) and X1 (next Dd rows), ld = Nn
// XMODE 2: line gather: k<Dd  -> X0[:, idx[ng]]   k<2Dd -> X0[:, idx[ng^1]]
//          else X2[:, ng]
// EPI   0: + bias                         (f32 store)
// EPI   1: relu((.+bias)*gamma + beta)    (f32 store)
// EPI   2: + bias + res                   (f32 store, residual)
// EPI   3: + bias, atomicAdd outp[:, idx[ng]]  (fused scatter-add)
// EPI   4: + bias, pack bf16 [b, h, n, dh]     (q/k for flash;  H=4, DH=64)
// EPI   5: + bias, pack bf16 [b, h, dh, n]     (v for flash)
// Tiles: block = 64(M) x 128(N), 8 waves, wave = 16x64, K-step 32.
// ---------------------------------------------------------------------------
template <int XMODE, int EPI>
__global__ void __launch_bounds__(256) gemm_wmma(
    const float* __restrict__ W, const float* __restrict__ bias,
    const float* __restrict__ gamma, const float* __restrict__ beta,
    const float* __restrict__ X0, const float* __restrict__ X1,
    const float* __restrict__ X2, const int* __restrict__ idxp,
    const float* __restrict__ res, float* __restrict__ outp,
    int M, int K, int Nc, int ldx0, int col_off,
    long x0_bs, long x1_bs, long x2_bs, long res_bs, long out_bs,
    int Dd, int Nn, int Ll, int ldout, int Hh, int DHh)
{
  __shared__ __align__(16) bf16 Wt[64][40];    // [m][k]   row = 80B (16B mult)
  __shared__ __align__(16) bf16 Xt[128][40];   // [n][k]   transposed for B-frags

  const int t    = threadIdx.x;
  const int wv   = t >> 5;
  const int lane = t & 31;
  const int l15  = lane & 15;
  const int b    = blockIdx.z;
  const int m0   = blockIdx.y * 64;
  const int n0   = blockIdx.x * 128;
  const int wm   = (wv & 3) * 16;    // wave M offset in tile
  const int wn   = (wv >> 2) * 64;   // wave N offset in tile
  const int kbA  = (lane >> 4) * 8;  // A K-interleave pattern
  const int kbB  = (lane >> 4) * 16; // unused; B fragments use the kbA order, same as A

  v8f acc[4] = {};

  for (int k0 = 0; k0 < K; k0 += 32) {
    // ---- stage W tile (64x32): float4 loads, one v8bf LDS store ----
    {
      const int row = t >> 2;
      const int kk  = (t & 3) * 8;
      const float* wp = &W[(long)(m0 + row) * K + (k0 + kk)];
      if (k0 + 32 < K) __builtin_prefetch(wp + 32, 0, 1);
      const float4 w0 = *(const float4*)wp;
      const float4 w1 = *(const float4*)(wp + 4);
      v8bf pk;
      pk[0] = (bf16)w0.x; pk[1] = (bf16)w0.y; pk[2] = (bf16)w0.z; pk[3] = (bf16)w0.w;
      pk[4] = (bf16)w1.x; pk[5] = (bf16)w1.y; pk[6] = (bf16)w1.z; pk[7] = (bf16)w1.w;
      *(v8bf*)&Wt[row][kk] = pk;
    }
    // ---- stage X tile (32k x 128n) transposed: two k-rows packed per b32 ----
#pragma unroll
    for (int it = 0; it < 2; ++it) {
      const int ka  = ((t >> 5) + 8 * it) * 2;  // even k row: 0,2,..,30
      const int c0  = (t & 31) * 4;             // 0..124
      const int kA0 = k0 + ka;
      const int kA1 = k0 + ka + 1;
      float r0[4], r1[4];
      if (XMODE == 0 || XMODE == 1) {
        const float *s0p, *s1p;
        if (XMODE == 0) {
          s0p = &X0[b * x0_bs + (long)kA0 * ldx0];
          s1p = &X0[b * x0_bs + (long)kA1 * ldx0];
        } else {
          s0p = (kA0 < Dd) ? &X0[b * x0_bs + (long)kA0 * Nn]
                           : &X1[b * x1_bs + (long)(kA0 - Dd) * Nn];
          s1p = (kA1 < Dd) ? &X0[b * x0_bs + (long)kA1 * Nn]
                           : &X1[b * x1_bs + (long)(kA1 - Dd) * Nn];
        }
        const float4 xa = *(const float4*)(s0p + n0 + c0);
        const float4 xb = *(const float4*)(s1p + n0 + c0);
        r0[0] = xa.x; r0[1] = xa.y; r0[2] = xa.z; r0[3] = xa.w;
        r1[0] = xb.x; r1[1] = xb.y; r1[2] = xb.z; r1[3] = xb.w;
      } else {
#pragma unroll
        for (int j = 0; j < 4; ++j) {
          const int ng = n0 + c0 + j + col_off;
#pragma unroll
          for (int hc = 0; hc < 2; ++hc) {
            const int kk = (hc == 0) ? kA0 : kA1;
            float xv;
            if (kk < 2 * Dd) {
              const int l  = (kk < Dd) ? ng : (ng ^ 1);
              const int kr = (kk < Dd) ? kk : (kk - Dd);
              const int jj = idxp[(long)b * Ll + l];
              xv = X0[b * x0_bs + (long)kr * ldx0 + jj];
            } else {
              xv = X2[b * x2_bs + (long)(kk - 2 * Dd) * Ll + ng];
            }
            if (hc == 0) r0[j] = xv; else r1[j] = xv;
          }
        }
      }
#pragma unroll
      for (int j = 0; j < 4; ++j) {
        v2bf pr;
        pr[0] = (bf16)r0[j];
        pr[1] = (bf16)r1[j];
        *(v2bf*)&Xt[c0 + j][ka] = pr;   // packed b32 store (k, k+1)
      }
    }
    __syncthreads();

    // ---- fragments: contiguous 16B LDS loads ----
    const int ar = wm + l15;
    const v16bf fa = cat8(ld8(&Wt[ar][kbA]), ld8(&Wt[ar][16 + kbA]));
#pragma unroll
    for (int i = 0; i < 4; ++i) {
      const int cn = wn + i * 16 + l15;
      const v16bf fb = cat8(ld8(&Xt[cn][kbA]), ld8(&Xt[cn][kbA + 16]));
      acc[i] = wmma_bf16(fa, fb, acc[i]);
    }
    __syncthreads();
  }

  // ---- epilogue: C layout lane<16:(M=r,N=lane) lane>=16:(M=8+r,N=lane-16) --
  const int mbase = m0 + wm + ((lane < 16) ? 0 : 8);
#pragma unroll
  for (int r = 0; r < 8; ++r) {
    const int m  = mbase + r;
    const float bz = bias[m];
#pragma unroll
    for (int i = 0; i < 4; ++i) {
      const int n = n0 + wn + i * 16 + l15;
      float vA = acc[i][r] + bz;
      if (EPI == 1) {
        vA = fmaxf(vA * gamma[m] + beta[m], 0.0f);
      }
      if (EPI == 2) {
        vA += res[b * res_bs + (long)m * ldout + n];
      }
      if (EPI == 3) {
        const int jA = idxp[(long)b * Ll + (n + col_off)];
        atomicAdd(&outp[b * out_bs + (long)m * Nn + jA], vA);
      } else if (EPI == 4) {          // pack [b, h, n, dh] bf16 (H=4)
        const int h = m & 3, dh = m >> 2;
        bf16* dst = (bf16*)outp;
        dst[((long)(b * Hh + h) * Nn + n) * DHh + dh] = (bf16)vA;
      } else if (EPI == 5) {          // pack [b, h, dh, n] bf16
        const int h = m & 3, dh = m >> 2;
        bf16* dst = (bf16*)outp;
        dst[((long)(b * Hh + h) * DHh + dh) * Nn + n] = (bf16)vA;
      } else {
        outp[b * out_bs + (long)m * ldout + n] = vA;
      }
    }
  }
}

// ---------------------------------------------------------------------------
// Flash attention: one wave per 16-query tile per (b, h). DH=64, H=4.
// Packed bf16 inputs: q,k = [b,h,n,dh], v = [b,h,dh,n] -> every fragment is
// two contiguous 16B global loads. 64-key tiles; online softmax with the
// row-SUM computed on the matrix pipe (wmma(P, ones)); only the row-max needs
// lane shuffles. All fragment addresses are strength-reduced: per-lane base
// pointers advance by constant byte strides, no multiplies in the hot loop.
// ---------------------------------------------------------------------------
__global__ void __launch_bounds__(128) flash_attn(
    const bf16* __restrict__ qp, const bf16* __restrict__ kp,
    const bf16* __restrict__ vp, float* __restrict__ out,
    int Hh, int Nn, int Dd, int DHh)
{
  __shared__ __align__(16) bf16 pT[4][16][72];   // per-wave P [query][64 keys]

  const int wv   = threadIdx.x >> 5;
  const int lane = threadIdx.x & 31;
  const int l15  = lane & 15;
  const int gid  = blockIdx.x * 4 + wv;
  const int qtiles = Nn >> 4;
  const int q0 = (gid % qtiles) * 16;
  const int h  = (gid / qtiles) % Hh;
  const int b  = gid / (qtiles * Hh);
  const long plane = (long)Dd * Nn;
  const float scale = 0.125f;   // 1/sqrt(64)
  const int kbA = (lane >> 4) * 8;
  const int kbB = (lane >> 4) * 16;
  const long bh = (long)(b * Hh + h);

  // Q^T A-fragments (row = query, K = dh), kept in registers (DH=64 -> 2)
  const bf16* qb0 = qp + (bh * Nn + (q0 + l15)) * DHh;
  const v16bf aq0 = cat8(ld8(qb0 + kbA),      ld8(qb0 + 16 + kbA));
  const v16bf aq1 = cat8(ld8(qb0 + 32 + kbA), ld8(qb0 + 48 + kbA));

  v16bf ones;
#pragma unroll
  for (int j = 0; j < 16; ++j) ones[j] = (bf16)1.0f;

  float mrun[8], lrun[8];
#pragma unroll
  for (int r = 0; r < 8; ++r) { mrun[r] = -1e30f; lrun[r] = 0.0f; }
  v8f o[4] = {};   // O tile: 16 queries x 64 dh

  // Loop-invariant strides (hoisted by LICM, no in-loop v_mul_u64).
  const long kSub   = (long)16 * DHh;     // 16 keys in [b,h,n,dh]
  const long kStep  = (long)64 * DHh;     // 64 keys per tile
  const long vRowT  = (long)16 * Nn;      // 16 dh rows in [b,h,dh,n]
  const bf16* kb = kp + (bh * Nn + l15) * DHh;   // key l15's dh vector
  const bf16* vb = vp + (bh * DHh + l15) * Nn;   // dh row l15, advances by keys

  for (int kt = 0; kt < Nn; kt += 64) {
    // ---- scores: 4 sub-tiles of 16 keys, K-dim = dh ----
    v8f s[4];
#pragma unroll
    for (int sub = 0; sub < 4; ++sub) {
      const bf16* kr = kb + sub * kSub;
      const v16bf f0 = cat8(ld8(kr + kbA),      ld8(kr + kbA + 16));
      const v16bf f1 = cat8(ld8(kr + 32 + kbA), ld8(kr + 48 + kbA));
      v8f a = {};
      a = wmma_bf16(aq0, f0, a);
      a = wmma_bf16(aq1, f1, a);
      s[sub] = a;
    }

    // ---- row max (shuffles) + exp + stage P to LDS ----
    float corr8[8];
#pragma unroll
    for (int r = 0; r < 8; ++r) {
      const float a0 = s[0][r] * scale;
      const float a1 = s[1][r] * scale;
      const float a2 = s[2][r] * scale;
      const float a3 = s[3][r] * scale;
      float rm = fmaxf(fmaxf(a0, a1), fmaxf(a2, a3));
      rm = fmaxf(rm, __shfl_xor(rm, 1, 32));
      rm = fmaxf(rm, __shfl_xor(rm, 2, 32));
      rm = fmaxf(rm, __shfl_xor(rm, 4, 32));
      rm = fmaxf(rm, __shfl_xor(rm, 8, 32));
      const float mn = fmaxf(mrun[r], rm);
      corr8[r] = __expf(mrun[r] - mn);
      mrun[r] = mn;
#pragma unroll
      for (int tt = 0; tt < 4; ++tt) o[tt][r] *= corr8[r];
      const int qr = (lane < 16) ? r : (8 + r);
      pT[wv][qr][l15]      = (bf16)__expf(a0 - mn);
      pT[wv][qr][16 + l15] = (bf16)__expf(a1 - mn);
      pT[wv][qr][32 + l15] = (bf16)__expf(a2 - mn);
      pT[wv][qr][48 + l15] = (bf16)__expf(a3 - mn);
    }
    asm volatile("s_wait_dscnt 0" ::: "memory");

    // ---- P as A-fragments (16 queries x 2x32 keys), 16B LDS loads ----
    const v16bf ap0 = cat8(ld8(&pT[wv][l15][kbA]),      ld8(&pT[wv][l15][16 + kbA]));
    const v16bf ap1 = cat8(ld8(&pT[wv][l15][32 + kbA]), ld8(&pT[wv][l15][48 + kbA]));

    // ---- row sums on the matrix pipe: sm[r] = sum_k P[row, k] ----
    v8f sm = {};
    sm = wmma_bf16(ap0, ones, sm);
    sm = wmma_bf16(ap1, ones, sm);
#pragma unroll
    for (int r = 0; r < 8; ++r) lrun[r] = lrun[r] * corr8[r] + sm[r];

    // ---- O += P @ V^T : B-fragments of V^T (K-dim = key, cols = dh) ----
#pragma unroll
    for (int tt = 0; tt < 4; ++tt) {
      const bf16* vr = vb + tt * vRowT;
      const v16bf fv0 = cat8(ld8(vr + kbA),      ld8(vr + kbA + 16));
      const v16bf fv1 = cat8(ld8(vr + 32 + kbA), ld8(vr + 48 + kbA));
      o[tt] = wmma_bf16(ap0, fv0, o[tt]);
      o[tt] = wmma_bf16(ap1, fv1, o[tt]);
    }
    kb += kStep;   // next 64 keys
    vb += 64;
  }

  // ---- normalize and store back to merged head-interleaved f32 [B,D,N] ----
#pragma unroll
  for (int r = 0; r < 8; ++r) {
    const float inv = 1.0f / lrun[r];
    const int qr = (lane < 16) ? r : (8 + r);
#pragma unroll
    for (int tt = 0; tt < 4; ++tt) {
      const int dh = tt * 16 + l15;
      out[b * plane + (long)(dh * Hh + h) * Nn + (q0 + qr)] = o[tt][r] * inv;
    }
  }
}

// ---------------------------------------------------------------------------
__global__ void count_kernel(const int* __restrict__ idx, float* __restrict__ cnt,
                             int Bb, int Ll, int Nn)
{
  const int i = blockIdx.x * blockDim.x + threadIdx.x;
  if (i < Bb * Ll) {
    const int b = i / Ll;
    atomicAdd(&cnt[b * Nn + idx[i]], 1.0f);
  }
}

__global__ void finalize_kernel(float* __restrict__ outD,
                                const float* __restrict__ sums,
                                const float* __restrict__ cnt,
                                int Bb, int Dd, int Nn)
{
  const long i = (long)blockIdx.x * blockDim.x + threadIdx.x;
  const long tot = (long)Bb * Dd * Nn;
  if (i < tot) {
    const int n  = (int)(i % Nn);
    const int bd = (int)(i / Nn);
    const int b  = bd / Dd;
    const float c = cnt[b * Nn + n];
    if (c > 0.0f) outD[i] += sums[i] / c;
  }
}

// ---------------------------------------------------------------------------
extern "C" void kernel_launch(void* const* d_in, const int* in_sizes, int n_in,
                              void* d_out, int out_size, void* d_ws, size_t ws_size,
                              hipStream_t stream)
{
  (void)in_sizes; (void)n_in; (void)out_size;
  const int Bv = 2, Dv = 256, Nv = 2048, Lv = 16384, Hv = 4, DHv = 64;

  const float* desc[2] = {(const float*)d_in[0], (const float*)d_in[1]};
  const float* lenc[2] = {(const float*)d_in[2], (const float*)d_in[3]};
  const float* Wq = (const float*)d_in[4];  const float* bq = (const float*)d_in[5];
  const float* Wk = (const float*)d_in[6];  const float* bk = (const float*)d_in[7];
  const float* Wv = (const float*)d_in[8];  const float* bv = (const float*)d_in[9];
  const float* Wm = (const float*)d_in[10]; const float* bm = (const float*)d_in[11];
  const float* aW1 = (const float*)d_in[12]; const float* ab1 = (const float*)d_in[13];
  const float* ag1 = (const float*)d_in[14]; const float* abe1 = (const float*)d_in[15];
  const float* aW2 = (const float*)d_in[16]; const float* ab2 = (const float*)d_in[17];
  const float* lW1 = (const float*)d_in[18]; const float* lb1 = (const float*)d_in[19];
  const float* lg1 = (const float*)d_in[20]; const float* lbe1 = (const float*)d_in[21];
  const float* lW2 = (const float*)d_in[22]; const float* lb2 = (const float*)d_in[23];
  const int* idx[2] = {(const int*)d_in[24], (const int*)d_in[25]};

  float* out = (float*)d_out;
  float* ws  = (float*)d_ws;

  const size_t planeDN = (size_t)Bv * Dv * Nv;           // 1M floats
  const size_t packE   = (size_t)Bv * Hv * Nv * DHv;     // bf16 elements (1M)
  size_t off = 0;
  auto alloc = [&](size_t n) { float* p = ws + off; off += n; return p; };
  bf16* qpack = (bf16*)alloc(packE / 2);
  bf16* kpack = (bf16*)alloc(packE / 2);
  bf16* vpack = (bf16*)alloc(packE / 2);
  float* attn = alloc(planeDN);
  float* msg  = alloc(planeDN);
  float* hA   = alloc((size_t)Bv * 2 * Dv * Nv);
  float* sums = alloc(planeDN);
  float* cnt  = alloc((size_t)Bv * Nv);

  // Adaptive line-chunk so the h-buffer fits remaining workspace.
  const size_t avail = ws_size / sizeof(float);
  int CH = 4096;
  while (CH > 128 && off + (size_t)Bv * 2 * Dv * CH > avail) CH >>= 1;
  float* hL = alloc((size_t)Bv * 2 * Dv * CH);

  const long dn = (long)Dv * Nv;
  const dim3 blk(256);

  for (int img = 0; img < 2; ++img) {
    const float* x  = desc[img];
    const float* le = lenc[img];
    const int* ix   = idx[img];
    float* outD     = out + (size_t)img * planeDN;

    // ---- q, k, v projections (pack bf16 for flash) ----
    const dim3 gQ(Nv / 128, Dv / 64, Bv);
    gemm_wmma<0, 4><<<gQ, blk, 0, stream>>>(Wq, bq, nullptr, nullptr,
        x, nullptr, nullptr, nullptr, nullptr, (float*)qpack,
        Dv, Dv, Nv, Nv, 0, dn, 0, 0, 0, 0, Dv, Nv, Lv, Nv, Hv, DHv);
    gemm_wmma<0, 4><<<gQ, blk, 0, stream>>>(Wk, bk, nullptr, nullptr,
        x, nullptr, nullptr, nullptr, nullptr, (float*)kpack,
        Dv, Dv, Nv, Nv, 0, dn, 0, 0, 0, 0, Dv, Nv, Lv, Nv, Hv, DHv);
    gemm_wmma<0, 5><<<gQ, blk, 0, stream>>>(Wv, bv, nullptr, nullptr,
        x, nullptr, nullptr, nullptr, nullptr, (float*)vpack,
        Dv, Dv, Nv, Nv, 0, dn, 0, 0, 0, 0, Dv, Nv, Lv, Nv, Hv, DHv);

    // ---- multi-head flash attention ----
    flash_attn<<<dim3(Bv * Hv * (Nv / 16) / 4), dim3(128), 0, stream>>>(
        qpack, kpack, vpack, attn, Hv, Nv, Dv, DHv);

    // ---- merge projection ----
    gemm_wmma<0, 0><<<gQ, blk, 0, stream>>>(Wm, bm, nullptr, nullptr,
        attn, nullptr, nullptr, nullptr, nullptr, msg,
        Dv, Dv, Nv, Nv, 0, dn, 0, 0, 0, dn, Dv, Nv, Lv, Nv, Hv, DHv);

    // ---- attention MLP: h = relu(BN(W1 @ [x; msg] + b1)) ----
    gemm_wmma<1, 1><<<dim3(Nv / 128, (2 * Dv) / 64, Bv), blk, 0, stream>>>(
        aW1, ab1, ag1, abe1, x, msg, nullptr, nullptr, nullptr, hA,
        2 * Dv, 2 * Dv, Nv, Nv, 0, dn, dn, 0, 0, 2 * dn, Dv, Nv, Lv, Nv, Hv, DHv);
    // ---- desc_mid = x + (W2 @ h + b2)  -> written into d_out ----
    gemm_wmma<0, 2><<<gQ, blk, 0, stream>>>(
        aW2, ab2, nullptr, nullptr, hA, nullptr, nullptr, nullptr, x, outD,
        Dv, 2 * Dv, Nv, Nv, 0, 2 * dn, 0, 0, dn, dn, Dv, Nv, Lv, Nv, Hv, DHv);

    // ---- line layer: gather -> MLP -> fused scatter-add ----
    hipMemsetAsync(sums, 0, planeDN * sizeof(float), stream);
    hipMemsetAsync(cnt, 0, (size_t)Bv * Nv * sizeof(float), stream);
    count_kernel<<<(Bv * Lv + 255) / 256, 256, 0, stream>>>(ix, cnt, Bv, Lv, Nv);

    for (int c0 = 0; c0 < Lv; c0 += CH) {
      gemm_wmma<2, 1><<<dim3(CH / 128, (2 * Dv) / 64, Bv), blk, 0, stream>>>(
          lW1, lb1, lg1, lbe1, outD, nullptr, le, ix, nullptr, hL,
          2 * Dv, 3 * Dv, CH, Nv, c0, dn, 0, (long)Dv * Lv, 0,
          (long)2 * Dv * CH, Dv, Nv, Lv, CH, Hv, DHv);
      gemm_wmma<0, 3><<<dim3(CH / 128, Dv / 64, Bv), blk, 0, stream>>>(
          lW2, lb2, nullptr, nullptr, hL, nullptr, nullptr, ix, nullptr, sums,
          Dv, 2 * Dv, CH, CH, c0, (long)2 * Dv * CH, 0, 0, 0,
          dn, Dv, Nv, Lv, CH, Hv, DHv);
    }

    finalize_kernel<<<(int)((planeDN + 255) / 256), 256, 0, stream>>>(
        outD, sums, cnt, Bv, Dv, Nv);
  }
}
